// Qwen3SeqParallelAttention_27951647162500
// MI455X (gfx1250) — compile-verified, matrix operand order checked by reading, not run on hardware
//
#include <hip/hip_runtime.h>
#include <hip/hip_bf16.h>

// ---- problem constants (from reference) ----
#define B_    4
#define S_    1024
#define HID_  4096
#define NH_   32
#define NKV_  8
#define HD_   128
#define QKVN_ 6144   // NH*HD + 2*NKV*HD

typedef unsigned short bfu;   // raw bf16 storage
typedef __attribute__((ext_vector_type(16))) __bf16 bf16x16;
typedef __attribute__((ext_vector_type(8)))  float  f32x8;
typedef __attribute__((ext_vector_type(4)))  unsigned int u32x4;
typedef __attribute__((ext_vector_type(8)))  int    i32x8;
typedef __attribute__((ext_vector_type(4)))  int    i32x4;

union Frag {       // one WMMA 16-bit A/B operand: 8 VGPRs = 32 bytes
  bf16x16 v;
  uint4   q[2];
};

__device__ __forceinline__ bfu f2bf(float f) {
  unsigned u = __float_as_uint(f);
  unsigned r = (u + 0x7fffu + ((u >> 16) & 1u)) >> 16;   // RNE
  return (bfu)r;
}

__device__ __forceinline__ f32x8 zero8() {
  f32x8 z;
#pragma unroll
  for (int i = 0; i < 8; ++i) z[i] = 0.f;
  return z;
}

__device__ __forceinline__ f32x8 wmma_bf16(const Frag& a, const Frag& b, f32x8 c) {
  // D(16x16,f32) = A(16x32,bf16) x B(32x16,bf16) + C
  return __builtin_amdgcn_wmma_f32_16x16x32_bf16(false, a.v, false, b.v,
                                                 (short)0, c, false, false);
}

// ---------------------------------------------------------------------------
// Tensor Data Mover: load a 128-row x 32-col bf16 tile (row stride = K elems)
// from global into LDS (rows packed contiguously, 32 elems each).
// D# layout per CDNA5 ISA 8.3/8.4: 2D tensor, groups 2/3 unused (zero).
// ---------------------------------------------------------------------------
__device__ __forceinline__ void tdm_load_tile_128x32(const bfu* gptr, bfu* lptr,
                                                     int K, int nrows) {
  const unsigned long long ga = (unsigned long long)gptr;
  const unsigned lds = (unsigned)(unsigned long long)lptr;  // LDS byte address

  u32x4 g0;
  g0[0] = 1u;                                   // count=1, user descriptor
  g0[1] = lds;                                  // lds_addr [63:32]
  g0[2] = (unsigned)ga;                         // global_addr lo
  g0[3] = (unsigned)((ga >> 32) & 0x01FFFFFFu)  // global_addr [56:32]
          | 0x80000000u;                        // type=2 ("image") [127:126]

  const unsigned uk = (unsigned)K, ur = (unsigned)nrows;
  i32x8 g1;
  g1[0] = 0x00010000;                           // data_size=1 (2 bytes); mask=0
  g1[1] = (int)(uk << 16);                      // [63:48] tensor_dim0 lo16
  g1[2] = (int)((uk >> 16) | (ur << 16));       // [79:64] dim0 hi, [95:80] dim1 lo
  g1[3] = (int)((ur >> 16) | (32u << 16));      // [111:96] dim1 hi, [127:112] tile_dim0=32
  g1[4] = 128;                                  // [143:128] tile_dim1=128 rows
  g1[5] = (int)uk;                              // [191:160] tensor_dim0_stride lo32
  g1[6] = 0;                                    // stride hi16 | dim1_stride lo16
  g1[7] = 0;

  i32x4 z4;
  z4[0] = 0; z4[1] = 0; z4[2] = 0; z4[3] = 0;

#if __clang_major__ >= 23
  i32x8 z8;
#pragma unroll
  for (int i = 0; i < 8; ++i) z8[i] = 0;
  __builtin_amdgcn_tensor_load_to_lds(g0, g1, z4, z4, z8, 0);
#else
  __builtin_amdgcn_tensor_load_to_lds(g0, g1, z4, z4, 0);
#endif
}

// ---------------------------------------------------------------------------
// Kernel 1: f32 -> bf16 cast (vectorized, grid-stride)
// ---------------------------------------------------------------------------
__global__ void cvt_f32_bf16(const float* __restrict__ src, bfu* __restrict__ dst,
                             size_t n4) {
  size_t i = (size_t)blockIdx.x * blockDim.x + threadIdx.x;
  size_t stride = (size_t)gridDim.x * blockDim.x;
  for (; i < n4; i += stride) {
    float4 f = ((const float4*)src)[i];
    ushort4 o;
    o.x = f2bf(f.x); o.y = f2bf(f.y); o.z = f2bf(f.z); o.w = f2bf(f.w);
    ((ushort4*)dst)[i] = o;
  }
}

// ---------------------------------------------------------------------------
// Kernel 2: C[M,N](f32) = A[M,K](bf16) * B[N,K]^T(bf16)
// 128x128 tile / WG, 8 waves, wave tile 32x64 (2x4 WMMA frags), K-step 32.
// Double-buffered TDM (tensor_load_to_lds) feeds LDS; waves only do WMMA.
// ---------------------------------------------------------------------------
__global__ __launch_bounds__(256) void gemm_bf16(const bfu* __restrict__ Ag,
                                                 const bfu* __restrict__ Bg,
                                                 float* __restrict__ C,
                                                 int M, int N, int K) {
  __shared__ bfu sA[2][128][32];   // 2 x 8 KB
  __shared__ bfu sB[2][128][32];   // 2 x 8 KB

  const int tid  = threadIdx.x;
  const int lane = tid & 31;
  const int half = lane >> 4;
  const int lr   = lane & 15;
  const int wave = tid >> 5;

  const int bm = blockIdx.y * 128;
  const int bn = blockIdx.x * 128;
  const int wm = (wave >> 1) * 32;   // wave row offset in tile
  const int wn = (wave & 1) * 64;    // wave col offset in tile

  // scalar (SGPR) wave id -> s_cbranch so the TDM issues exactly once
  const int swave = __builtin_amdgcn_readfirstlane(wave);

  f32x8 acc[2][4];
#pragma unroll
  for (int i = 0; i < 2; ++i)
#pragma unroll
    for (int j = 0; j < 4; ++j) acc[i][j] = zero8();

  // prologue: DMA first K-slice pair into buffer 0
  if (swave == 0) {
    tdm_load_tile_128x32(Ag + (size_t)bm * K, &sA[0][0][0], K, M);
    tdm_load_tile_128x32(Bg + (size_t)bn * K, &sB[0][0][0], K, N);
  }

  for (int k0 = 0; k0 < K; k0 += 32) {
    const int buf = (k0 >> 5) & 1;
    if (swave == 0) {
      if (k0 + 32 < K) {
        // DMA next slice into the other buffer, then wait for current pair.
        // TDM ops complete in order per wave: <=2 outstanding => current done.
        tdm_load_tile_128x32(Ag + (size_t)bm * K + k0 + 32, &sA[buf ^ 1][0][0], K, M);
        tdm_load_tile_128x32(Bg + (size_t)bn * K + k0 + 32, &sB[buf ^ 1][0][0], K, N);
        __builtin_amdgcn_s_wait_tensorcnt(2);
      } else {
        __builtin_amdgcn_s_wait_tensorcnt(0);
      }
    }
    __syncthreads();   // current buffer visible to all waves

    // build fragments (ISA 7.12.2: lane<16 -> K 0-7,16-23 ; lane>=16 -> K 8-15,24-31)
    Frag af[2], bf[4];
#pragma unroll
    for (int i = 0; i < 2; ++i) {
      const int row = wm + i * 16 + lr;
      af[i].q[0] = *(const uint4*)&sA[buf][row][half * 8];
      af[i].q[1] = *(const uint4*)&sA[buf][row][16 + half * 8];
    }
#pragma unroll
    for (int j = 0; j < 4; ++j) {
      const int col = wn + j * 16 + lr;
      bf[j].q[0] = *(const uint4*)&sB[buf][col][half * 8];
      bf[j].q[1] = *(const uint4*)&sB[buf][col][16 + half * 8];
    }
#pragma unroll
    for (int i = 0; i < 2; ++i)
#pragma unroll
      for (int j = 0; j < 4; ++j)
        acc[i][j] = wmma_bf16(af[i], bf[j], acc[i][j]);

    __syncthreads();   // done reading; next TDM may overwrite buf^1
  }

  // epilogue: C tile layout — vgpr r: M=r (lanes 0-15), M=8+r (lanes 16-31); N=lane%16
#pragma unroll
  for (int i = 0; i < 2; ++i)
#pragma unroll
    for (int j = 0; j < 4; ++j)
#pragma unroll
      for (int r = 0; r < 8; ++r) {
        const int row = bm + wm + i * 16 + r + 8 * half;
        const int col = bn + wn + j * 16 + lr;
        C[(size_t)row * N + col] = acc[i][j][r];
      }
}

// ---------------------------------------------------------------------------
// Kernel 3: per-head RMSNorm + RoPE on q/k, cast to bf16; pass-through v
// one wave per (token, head-slot); head-slots: 0-31 q, 32-39 k, 40-47 v
// ---------------------------------------------------------------------------
__global__ __launch_bounds__(256) void postqkv(const float* __restrict__ qkv,
                                               const int* __restrict__ positions,
                                               const float* __restrict__ qw,
                                               const float* __restrict__ kw,
                                               bfu* __restrict__ Qb,
                                               bfu* __restrict__ Kb,
                                               bfu* __restrict__ Vb) {
  const int gw   = (blockIdx.x * blockDim.x + threadIdx.x) >> 5;
  const int lane = threadIdx.x & 31;
  const int total = B_ * S_ * 48;
  if (gw >= total) return;

  const int j     = gw % 48;
  const int token = gw / 48;
  const int s     = token % S_;

  const float* src = qkv + (size_t)token * QKVN_ + (size_t)j * HD_;
  float x0 = src[lane], x1 = src[lane + 32], x2 = src[lane + 64], x3 = src[lane + 96];

  if (j < 40) {  // q or k: rmsnorm + rope
    float ss = x0 * x0 + x1 * x1 + x2 * x2 + x3 * x3;
#pragma unroll
    for (int off = 1; off < 32; off <<= 1) ss += __shfl_xor(ss, off, 32);
    const float scale = rsqrtf(ss * (1.0f / HD_) + 1e-6f);
    const float* w = (j < 32) ? qw : kw;
    x0 *= scale * w[lane];       x1 *= scale * w[lane + 32];
    x2 *= scale * w[lane + 64];  x3 *= scale * w[lane + 96];

    const float pos = (float)positions[s];
    // pairs (d, d+64): lane owns freq idx lane and lane+32
    const float inv0 = __powf(10000.f, -(float)(2 * lane) / 128.f);
    const float inv1 = __powf(10000.f, -(float)(2 * (lane + 32)) / 128.f);
    const float a0 = pos * inv0, a1 = pos * inv1;
    const float c0 = __cosf(a0), s0 = __sinf(a0);
    const float c1 = __cosf(a1), s1 = __sinf(a1);
    const float o0 = x0 * c0 - x2 * s0, o2 = x2 * c0 + x0 * s0;
    const float o1 = x1 * c1 - x3 * s1, o3 = x3 * c1 + x1 * s1;
    x0 = o0; x1 = o1; x2 = o2; x3 = o3;
  }

  bfu* dst;
  if (j < 32)      dst = Qb + ((size_t)token * NH_  + j)        * HD_;
  else if (j < 40) dst = Kb + ((size_t)token * NKV_ + (j - 32)) * HD_;
  else             dst = Vb + ((size_t)token * NKV_ + (j - 40)) * HD_;
  dst[lane]      = f2bf(x0); dst[lane + 32] = f2bf(x1);
  dst[lane + 64] = f2bf(x2); dst[lane + 96] = f2bf(x3);
}

// ---------------------------------------------------------------------------
// Kernel 4: flash attention (causal, GQA). grid = (S/128, NH, B), 8 waves/WG,
// each wave owns 16 query rows; K/V staged in LDS per 32-key block.
// ---------------------------------------------------------------------------
__global__ __launch_bounds__(256) void attn_flash(const bfu* __restrict__ Qb,
                                                  const bfu* __restrict__ Kb,
                                                  const bfu* __restrict__ Vb,
                                                  bfu* __restrict__ Ob) {
  __shared__ bfu sK[32][128];    // 8 KB: [key][d]
  __shared__ bfu sVt[128][32];   // 8 KB: [d][key] (transposed for B-frags)
  __shared__ bfu sP[8][16][32];  // 8 KB: per-wave P repack scratch

  const int b    = blockIdx.z;
  const int h    = blockIdx.y;
  const int qblk = blockIdx.x;
  const int kvh  = h >> 2;       // NH/NKV = 4

  const int tid  = threadIdx.x;
  const int wave = tid >> 5;
  const int lane = tid & 31;
  const int half = lane >> 4;
  const int lr   = lane & 15;

  const int qbase = qblk * 128 + wave * 16;

  // Q fragments for this wave's 16 rows (4 A-frags cover HD=128)
  Frag qf[4];
  {
    const bfu* Qrow = Qb + ((size_t)(b * S_ + qbase + lr) * NH_ + h) * HD_;
#pragma unroll
    for (int kk = 0; kk < 4; ++kk) {
      qf[kk].q[0] = *(const uint4*)(Qrow + kk * 32 + half * 8);
      qf[kk].q[1] = *(const uint4*)(Qrow + kk * 32 + 16 + half * 8);
    }
  }

  f32x8 acc[8];
  float rmax[8], rsum[8];
#pragma unroll
  for (int dt = 0; dt < 8; ++dt) acc[dt] = zero8();
#pragma unroll
  for (int r = 0; r < 8; ++r) { rmax[r] = -1e30f; rsum[r] = 0.f; }

  const int nkb = (qblk + 1) * 4;   // 32-key blocks up to end of this q block
  const float scale = 0.08838834764831845f;  // 1/sqrt(128)

  for (int kb = 0; kb < nkb; ++kb) {
    const int kbase = kb * 32;
    // cooperative stage: K row-major, V transposed
    {
      const int key = tid >> 3;          // 0..31
      const int c   = (tid & 7) * 16;    // 0..112
      const uint4* ks = (const uint4*)(Kb + ((size_t)(b * S_ + kbase + key) * NKV_ + kvh) * HD_ + c);
      *(uint4*)&sK[key][c]     = ks[0];
      *(uint4*)&sK[key][c + 8] = ks[1];
      const bfu* vs = Vb + ((size_t)(b * S_ + kbase + key) * NKV_ + kvh) * HD_ + c;
#pragma unroll
      for (int e = 0; e < 16; ++e) sVt[c + e][key] = vs[e];
    }
    __syncthreads();

    // S = Q*K^T for two 16-key halves
    f32x8 st[2];
#pragma unroll
    for (int hh = 0; hh < 2; ++hh) {
      f32x8 s = zero8();
      const int key = hh * 16 + lr;
#pragma unroll
      for (int kk = 0; kk < 4; ++kk) {
        Frag kf;
        kf.q[0] = *(const uint4*)&sK[key][kk * 32 + half * 8];
        kf.q[1] = *(const uint4*)&sK[key][kk * 32 + 16 + half * 8];
        s = wmma_bf16(qf[kk], kf, s);
      }
      st[hh] = s;
    }

    // online softmax update (row stats live per vgpr index; width-16 reductions)
#pragma unroll
    for (int r = 0; r < 8; ++r) {
      const int qi = qbase + r + 8 * half;
#pragma unroll
      for (int hh = 0; hh < 2; ++hh) {
        const int ki = kbase + hh * 16 + lr;
        float v = st[hh][r] * scale;
        if (ki > qi) v = -1e30f;
        st[hh][r] = v;
      }
      float m = fmaxf(st[0][r], st[1][r]);
#pragma unroll
      for (int off = 1; off < 16; off <<= 1) m = fmaxf(m, __shfl_xor(m, off, 16));
      const float nmax  = fmaxf(rmax[r], m);
      const float alpha = __expf(rmax[r] - nmax);
      rmax[r] = nmax;
      const float p0 = __expf(st[0][r] - nmax);
      const float p1 = __expf(st[1][r] - nmax);
      st[0][r] = p0; st[1][r] = p1;
      float sum = p0 + p1;
#pragma unroll
      for (int off = 1; off < 16; off <<= 1) sum += __shfl_xor(sum, off, 16);
      rsum[r] = rsum[r] * alpha + sum;
#pragma unroll
      for (int dt = 0; dt < 8; ++dt) acc[dt][r] *= alpha;
    }

    // repack P (C layout) -> A-frag layout via wave-private LDS bounce
#pragma unroll
    for (int r = 0; r < 8; ++r) {
      const int row = r + 8 * half;
      sP[wave][row][lr]      = f2bf(st[0][r]);
      sP[wave][row][16 + lr] = f2bf(st[1][r]);
    }
    Frag pf;   // wave-internal RAW through LDS; DS ops in-order per wave
    pf.q[0] = *(const uint4*)&sP[wave][lr][half * 8];
    pf.q[1] = *(const uint4*)&sP[wave][lr][16 + half * 8];

    // O += P * V  (8 d-tiles)
#pragma unroll
    for (int dt = 0; dt < 8; ++dt) {
      Frag vf;
      const int d = dt * 16 + lr;
      vf.q[0] = *(const uint4*)&sVt[d][half * 8];
      vf.q[1] = *(const uint4*)&sVt[d][16 + half * 8];
      acc[dt] = wmma_bf16(pf, vf, acc[dt]);
    }
    __syncthreads();
  }

  // epilogue: normalize, write bf16 attention output (token, h*HD + d)
#pragma unroll
  for (int dt = 0; dt < 8; ++dt)
#pragma unroll
    for (int r = 0; r < 8; ++r) {
      const int row = qbase + r + 8 * half;
      const float o = acc[dt][r] / rsum[r];
      Ob[(size_t)(b * S_ + row) * (NH_ * HD_) + h * HD_ + dt * 16 + lr] = f2bf(o);
    }
}

// ---------------------------------------------------------------------------
// launch
// ---------------------------------------------------------------------------
extern "C" void kernel_launch(void* const* d_in, const int* in_sizes, int n_in,
                              void* d_out, int out_size, void* d_ws, size_t ws_size,
                              hipStream_t stream) {
  (void)in_sizes; (void)n_in; (void)out_size; (void)ws_size;
  const float* h_shard   = (const float*)d_in[0];
  const int*   positions = (const int*)d_in[1];
  const float* w_qkv     = (const float*)d_in[2];
  const float* w_o       = (const float*)d_in[3];
  const float* q_norm_w  = (const float*)d_in[4];
  const float* k_norm_w  = (const float*)d_in[5];
  float* out = (float*)d_out;

  char* ws = (char*)d_ws;
  size_t off = 0;
  bfu* hb    = (bfu*)(ws + off); off += (size_t)B_ * S_ * HID_ * 2;     // 32 MiB
  bfu* wqkvb = (bfu*)(ws + off); off += (size_t)QKVN_ * HID_ * 2;       // 48 MiB
  bfu* wob   = (bfu*)(ws + off); off += (size_t)HID_ * HID_ * 2;        // 32 MiB
  bfu* Qb    = (bfu*)(ws + off); off += (size_t)B_ * S_ * NH_ * HD_ * 2;
  bfu* Kb    = (bfu*)(ws + off); off += (size_t)B_ * S_ * NKV_ * HD_ * 2;
  bfu* Vb    = (bfu*)(ws + off); off += (size_t)B_ * S_ * NKV_ * HD_ * 2;
  float* qkv = (float*)(ws + off); off += (size_t)B_ * S_ * QKVN_ * 4;  // 96 MiB
  bfu* attnb = (bfu*)qkv;  // alias: qkv f32 is dead after postqkv

  const int M = B_ * S_;  // 4096 tokens

  cvt_f32_bf16<<<2048, 256, 0, stream>>>(h_shard, hb,    (size_t)M * HID_ / 4);
  cvt_f32_bf16<<<2048, 256, 0, stream>>>(w_qkv,   wqkvb, (size_t)QKVN_ * HID_ / 4);
  cvt_f32_bf16<<<2048, 256, 0, stream>>>(w_o,     wob,   (size_t)HID_ * HID_ / 4);

  gemm_bf16<<<dim3(QKVN_ / 128, M / 128), 256, 0, stream>>>(hb, wqkvb, qkv, M, QKVN_, HID_);

  postqkv<<<(B_ * S_ * 48) / 8, 256, 0, stream>>>(qkv, positions, q_norm_w, k_norm_w,
                                                  Qb, Kb, Vb);

  attn_flash<<<dim3(S_ / 128, NH_, B_), 256, 0, stream>>>(Qb, Kb, Vb, attnb);

  gemm_bf16<<<dim3(HID_ / 128, M / 128), 256, 0, stream>>>(attnb, wob, out, M, HID_, HID_);
}
